// MultiHeadAttention_52742198394886
// MI455X (gfx1250) — compile-verified, matrix operand order checked by reading, not run on hardware
//
#include <hip/hip_runtime.h>
#include <hip/hip_bf16.h>

// ---------------------------------------------------------------------------
// Causal multi-head attention forward for MI455X (gfx1250), wave32 + WMMA.
// All matmuls run through v_wmma_f32_16x16x32_bf16 (f32 accumulate).
// QKV projection stages its block-shared weight tiles through LDS with
// GLOBAL_LOAD_ASYNC_TO_LDS_B128 (double-buffered, s_wait_asynccnt).
// ---------------------------------------------------------------------------

typedef __bf16 bf16_t;
typedef __attribute__((ext_vector_type(16))) __bf16 v16bf;
typedef __attribute__((ext_vector_type(8)))  __bf16 v8bf;
typedef __attribute__((ext_vector_type(8)))  float  v8f;

static constexpr int cB = 4;
static constexpr int cT = 2048;
static constexpr int cE = 1024;
static constexpr int cH = 16;
static constexpr int cD = 64;

// ---------------------------------------------------------------------------
// Fragment helpers (CDNA5 WMMA VGPR layouts, wave32).
//
// 16-bit A (16x32, MxK):  lanes 0-15 hold row M=lane,   K in {0..7, 16..23}
//                         lanes 16-31 hold row M=lane-16, K in {8..15, 24..31}
// B (32x16, KxN) loaded from N-major storage (i.e. B^T row-major) uses the
// identical per-lane pattern with N = lane&15.
// C/D (16x16 f32): VGPR r -> row r (lanes 0-15) / row r+8 (lanes 16-31),
//                  col = lane&15.
// ---------------------------------------------------------------------------
__device__ __forceinline__ v16bf load_frag16(const bf16_t* tile, int ld) {
  const int lane = threadIdx.x & 31;
  const int half = lane >> 4;
  const int r    = lane & 15;
  const bf16_t* p = tile + (size_t)r * ld + half * 8;
  v8bf lo = *(const v8bf*)(p);        // K = half*8 + 0..7      (16B load)
  v8bf hi = *(const v8bf*)(p + 16);   // K = half*8 + 16..23    (16B load)
  v16bf f;
#pragma unroll
  for (int e = 0; e < 8; ++e) { f[e] = lo[e]; f[8 + e] = hi[e]; }
  return f;
}

__device__ __forceinline__ v8f wmma_bf16(v16bf a, v16bf b, v8f c) {
  return __builtin_amdgcn_wmma_f32_16x16x32_bf16(
      /*neg_a=*/false, a, /*neg_b=*/false, b,
      /*c_mod=*/(short)0, c, /*reuse_a=*/false, /*reuse_b=*/false);
}

// ---------------------------------------------------------------------------
// Async global->LDS copy (CDNA5, tracked with ASYNCcnt).  16 bytes per lane.
// vdst VGPR carries the LDS byte address; flat pointers to __shared__ carry
// the addrspace(3) offset in their low 32 bits on amdgcn.
// ---------------------------------------------------------------------------
__device__ __forceinline__ void async_copy_b128(bf16_t* lds_dst, const bf16_t* gsrc) {
  unsigned int laddr = (unsigned int)(uintptr_t)lds_dst;
  asm volatile("global_load_async_to_lds_b128 %0, %1, off"
               :: "v"(laddr), "v"(gsrc)
               : "memory");
}

__device__ __forceinline__ void wait_asynccnt0() {
  asm volatile("s_wait_asynccnt 0" ::: "memory");
}

// ---------------------------------------------------------------------------
// Pack kernels: f32 -> bf16, plus weight transpose [H,E,D] -> [H,D,E].
// ---------------------------------------------------------------------------
__global__ void cvt_f32_bf16_kernel(bf16_t* __restrict__ dst,
                                    const float* __restrict__ src, int n) {
  int i = blockIdx.x * blockDim.x + threadIdx.x;
  if (i < n) dst[i] = (bf16_t)src[i];
}

__global__ void transpose_w_kernel(bf16_t* __restrict__ dst,
                                   const float* __restrict__ src) {
  // dst[h][d][e] = src[h][e][d]
  int i = blockIdx.x * blockDim.x + threadIdx.x;
  if (i >= cH * cD * cE) return;
  int h   = i / (cD * cE);
  int rem = i % (cD * cE);
  int d   = rem / cE;
  int e   = rem % cE;
  dst[i] = (bf16_t)src[((size_t)h * cE + e) * cD + d];
}

// ---------------------------------------------------------------------------
// QKV projection: out = x @ W  (per head).  One wave computes a 16x64 strip.
// All 8 waves of a block share (b,h) (16 blocks per head, never straddling),
// so the 64x32 weight tile of each K-chunk is staged once per block into a
// double-buffered LDS tile via async DMA and consumed by all waves.
//   xb : [B*T, E] bf16          Wt : [H, D, E] bf16 (pre-transposed)
//   transposed==0 -> out[b][h][t][d]  (Q, K)
//   transposed==1 -> out[b][h][d][t]  (V stored transposed for attn.V WMMA)
// ---------------------------------------------------------------------------
__device__ __forceinline__ void stage_w_tile(const bf16_t* wb, bf16_t* dst, int kc) {
  // 256 threads x 16 B = 4 KB = 64 rows x 32 bf16
  const int tid = threadIdx.x;
  const int d   = tid >> 2;
  const int kq  = (tid & 3) * 8;
  async_copy_b128(dst + d * 32 + kq, wb + (size_t)d * cE + kc + kq);
}

__global__ void __launch_bounds__(256)
qkv_kernel(const bf16_t* __restrict__ xb, const bf16_t* __restrict__ Wt,
           bf16_t* __restrict__ out, int transposed) {
  __shared__ __align__(16) bf16_t wstage[2][64 * 32];   // 8 KB double buffer

  const int lane  = threadIdx.x & 31;
  const int wave  = threadIdx.x >> 5;
  const int strip = blockIdx.x * 8 + wave;        // 0 .. B*H*(T/16)-1
  const int tilesT = cT / 16;
  const int bh = strip / tilesT;                  // block-uniform
  const int it = strip % tilesT;
  const int b = bh / cH, h = bh % cH;

  const bf16_t* xrow = xb + (size_t)(b * cT + it * 16) * cE;
  const bf16_t* wb   = Wt + (size_t)h * cD * cE;

  stage_w_tile(wb, &wstage[0][0], 0);             // prologue DMA

  v8f acc[4] = {};
  const int nchunks = cE / 32;
  for (int kc = 0; kc < nchunks; ++kc) {
    const int buf = kc & 1;
    wait_asynccnt0();                             // my part of stage kc landed
    __syncthreads();                              // everyone's part landed
    if (kc + 1 < nchunks)                         // overlap next DMA w/ math
      stage_w_tile(wb, &wstage[buf ^ 1][0], (kc + 1) * 32);

    __builtin_prefetch(xrow + kc * 32 + 64, 0, 1);
    v16bf a = load_frag16(xrow + kc * 32, cE);
#pragma unroll
    for (int n = 0; n < 4; ++n) {
      v16bf bf = load_frag16(&wstage[buf][n * 16 * 32], 32);
      acc[n] = wmma_bf16(a, bf, acc[n]);
    }
  }

  const int half = lane >> 4, col = lane & 15;
  if (!transposed) {
#pragma unroll
    for (int n = 0; n < 4; ++n)
#pragma unroll
      for (int r = 0; r < 8; ++r) {
        int t = it * 16 + r + half * 8;
        int d = n * 16 + col;
        out[(((size_t)b * cH + h) * cT + t) * cD + d] = (bf16_t)acc[n][r];
      }
  } else {
#pragma unroll
    for (int n = 0; n < 4; ++n)
#pragma unroll
      for (int r = 0; r < 8; ++r) {
        int t = it * 16 + r + half * 8;
        int d = n * 16 + col;
        out[(((size_t)b * cH + h) * cD + d) * cT + t] = (bf16_t)acc[n][r];
      }
  }
}

// ---------------------------------------------------------------------------
// Flash-style causal attention.  One wave owns one (b,h, 16-row query tile).
// Online softmax kept in the WMMA C layout; P goes f32->bf16 through a
// per-wave LDS tile to reshape into the WMMA A layout.  (No block barriers:
// waves have divergent causal trip counts; DS ops are in-order per wave.)
// ---------------------------------------------------------------------------
__global__ void __launch_bounds__(256)
attn_kernel(const bf16_t* __restrict__ Q, const bf16_t* __restrict__ Km,
            const bf16_t* __restrict__ Vt, bf16_t* __restrict__ O) {
  __shared__ __align__(16) bf16_t sP[8][16 * 32];   // per-wave 16x32 P tile

  const int lane  = threadIdx.x & 31;
  const int wave  = threadIdx.x >> 5;
  bf16_t* Pb = &sP[wave][0];

  const int strip = blockIdx.x * 8 + wave;          // 0 .. B*H*(T/16)-1
  const int tilesT = cT / 16;
  const int bh = strip / tilesT;
  const int iq = strip % tilesT;
  const int b = bh / cH, h = bh % cH;

  const bf16_t* Qb = Q  + (((size_t)b * cH + h) * cT + iq * 16) * cD;
  const bf16_t* Kb = Km + (((size_t)b * cH + h) * cT) * cD;
  const bf16_t* Vb = Vt + (((size_t)b * cH + h) * cD) * cT;

  const v16bf aq0 = load_frag16(Qb +  0, cD);
  const v16bf aq1 = load_frag16(Qb + 32, cD);

  v8f o[4] = {};
  float m[8], l[8];
#pragma unroll
  for (int r = 0; r < 8; ++r) { m[r] = -__builtin_inff(); l[r] = 0.f; }

  const float scale = 0.125f;                       // 1/sqrt(64)
  const int half = lane >> 4, cn = lane & 15;
  const int nkeys = iq * 16 + 16;                   // causal: keys 0..qmax

  for (int s0 = 0; s0 < nkeys; s0 += 32) {
    // ---- scores for two 16-key tiles -------------------------------------
    v8f st[2];
#pragma unroll
    for (int jj = 0; jj < 2; ++jj) {
      const int sbase = s0 + jj * 16;
      if (sbase < nkeys) {
        v8f c = {};
        v16bf bk0 = load_frag16(Kb + (size_t)sbase * cD +  0, cD);
        v16bf bk1 = load_frag16(Kb + (size_t)sbase * cD + 32, cD);
        c = wmma_bf16(aq0, bk0, c);
        c = wmma_bf16(aq1, bk1, c);
#pragma unroll
        for (int r = 0; r < 8; ++r) c[r] *= scale;
        if (sbase + 15 > iq * 16) {                 // tile touches diagonal
          const int s = sbase + cn;
#pragma unroll
          for (int r = 0; r < 8; ++r) {
            int trow = iq * 16 + r + half * 8;
            if (s > trow) c[r] = -__builtin_inff();
          }
        }
        st[jj] = c;
      } else {
#pragma unroll
        for (int r = 0; r < 8; ++r) st[jj][r] = -__builtin_inff();
      }
    }

    // ---- online softmax (row stats across each 16-lane half) -------------
    float mn[8];
#pragma unroll
    for (int r = 0; r < 8; ++r) mn[r] = fmaxf(m[r], fmaxf(st[0][r], st[1][r]));
#pragma unroll
    for (int off = 1; off < 16; off <<= 1)
#pragma unroll
      for (int r = 0; r < 8; ++r)
        mn[r] = fmaxf(mn[r], __shfl_xor(mn[r], off, 32));

    float alpha[8], rs[8];
#pragma unroll
    for (int r = 0; r < 8; ++r) {
      alpha[r] = __expf(m[r] - mn[r]);
      st[0][r] = __expf(st[0][r] - mn[r]);
      st[1][r] = __expf(st[1][r] - mn[r]);
      rs[r]    = st[0][r] + st[1][r];
      m[r]     = mn[r];
    }
#pragma unroll
    for (int off = 1; off < 16; off <<= 1)
#pragma unroll
      for (int r = 0; r < 8; ++r)
        rs[r] += __shfl_xor(rs[r], off, 32);
#pragma unroll
    for (int r = 0; r < 8; ++r) {
      l[r] = l[r] * alpha[r] + rs[r];
      o[0][r] *= alpha[r]; o[1][r] *= alpha[r];
      o[2][r] *= alpha[r]; o[3][r] *= alpha[r];
    }

    // ---- reshape P (C layout -> A layout) via LDS, as bf16 ---------------
#pragma unroll
    for (int jj = 0; jj < 2; ++jj)
#pragma unroll
      for (int r = 0; r < 8; ++r)
        Pb[(r + half * 8) * 32 + jj * 16 + cn] = (bf16_t)st[jj][r];
    // wave-local producer/consumer; DS ops are in-order per wave, make the
    // data dependency explicit and keep the compiler from reordering.
    asm volatile("s_wait_dscnt 0" ::: "memory");
    v16bf ap = load_frag16(Pb, 32);

    // ---- O += P @ V  (V stored transposed: contiguous in key dim) --------
#pragma unroll
    for (int n = 0; n < 4; ++n) {
      v16bf bv = load_frag16(Vb + (size_t)(n * 16) * cT + s0, cT);
      o[n] = wmma_bf16(ap, bv, o[n]);
    }
  }

  // ---- normalize and emit concat-head output [B*T, H*D] as bf16 ----------
  float inv[8];
#pragma unroll
  for (int r = 0; r < 8; ++r) inv[r] = 1.0f / l[r];
#pragma unroll
  for (int n = 0; n < 4; ++n)
#pragma unroll
    for (int r = 0; r < 8; ++r) {
      int t    = iq * 16 + r + half * 8;
      int ccol = h * cD + n * 16 + cn;
      O[((size_t)b * cT + t) * cE + ccol] = (bf16_t)(o[n][r] * inv[r]);
    }
}

// ---------------------------------------------------------------------------
// Output projection: y = O @ Wp^T + bp.  Wp row-major is already K-contiguous
// for the B fragment.  One wave computes a 16x64 strip, f32 output.
// ---------------------------------------------------------------------------
__global__ void __launch_bounds__(256)
proj_kernel(const bf16_t* __restrict__ O, const bf16_t* __restrict__ Wpb,
            const float* __restrict__ bp, float* __restrict__ y) {
  const int lane  = threadIdx.x & 31;
  const int wave  = threadIdx.x >> 5;
  const int strip = blockIdx.x * 8 + wave;          // 0 .. (B*T/16)*(E/64)-1
  const int ncolgrp = cE / 64;                      // 16
  const int it = strip / ncolgrp;                   // row tile
  const int ng = strip % ncolgrp;                   // 64-col group

  const bf16_t* Arow = O + (size_t)it * 16 * cE;

  v8f acc[4] = {};
  for (int kc = 0; kc < cE; kc += 32) {
    __builtin_prefetch(Arow + kc + 64, 0, 1);
    v16bf a = load_frag16(Arow + kc, cE);
#pragma unroll
    for (int n = 0; n < 4; ++n) {
      v16bf bf = load_frag16(Wpb + (size_t)(ng * 64 + n * 16) * cE + kc, cE);
      acc[n] = wmma_bf16(a, bf, acc[n]);
    }
  }

  const int half = lane >> 4, cn = lane & 15;
#pragma unroll
  for (int n = 0; n < 4; ++n) {
    const int ocol = ng * 64 + n * 16 + cn;
    const float bias = bp[ocol];
#pragma unroll
    for (int r = 0; r < 8; ++r) {
      int row = it * 16 + r + half * 8;
      y[(size_t)row * cE + ocol] = acc[n][r] + bias;
    }
  }
}

// ---------------------------------------------------------------------------
// Host launcher.
// ---------------------------------------------------------------------------
extern "C" void kernel_launch(void* const* d_in, const int* in_sizes, int n_in,
                              void* d_out, int out_size, void* d_ws, size_t ws_size,
                              hipStream_t stream) {
  (void)in_sizes; (void)n_in; (void)out_size; (void)ws_size;

  const float* x  = (const float*)d_in[0];   // [B,T,E]
  const float* Wq = (const float*)d_in[1];   // [H,E,D]
  const float* Wk = (const float*)d_in[2];   // [H,E,D]
  const float* Wv = (const float*)d_in[3];   // [H,E,D]
  const float* Wp = (const float*)d_in[4];   // [E,E]
  const float* bp = (const float*)d_in[5];   // [E]
  float* y = (float*)d_out;                  // [B,T,E]

  // workspace carve-out (bf16 scratch, ~88 MB total)
  char* ws = (char*)d_ws;
  bf16_t* xb  = (bf16_t*)ws; ws += (size_t)cB * cT * cE * sizeof(bf16_t);
  bf16_t* Wqt = (bf16_t*)ws; ws += (size_t)cH * cD * cE * sizeof(bf16_t);
  bf16_t* Wkt = (bf16_t*)ws; ws += (size_t)cH * cD * cE * sizeof(bf16_t);
  bf16_t* Wvt = (bf16_t*)ws; ws += (size_t)cH * cD * cE * sizeof(bf16_t);
  bf16_t* Wpb = (bf16_t*)ws; ws += (size_t)cE * cE * sizeof(bf16_t);
  bf16_t* Qb  = (bf16_t*)ws; ws += (size_t)cB * cH * cT * cD * sizeof(bf16_t);
  bf16_t* Kb  = (bf16_t*)ws; ws += (size_t)cB * cH * cT * cD * sizeof(bf16_t);
  bf16_t* Vtb = (bf16_t*)ws; ws += (size_t)cB * cH * cD * cT * sizeof(bf16_t);
  bf16_t* Ob  = (bf16_t*)ws; ws += (size_t)cB * cT * cE * sizeof(bf16_t);

  // 1) pack to bf16 (and transpose per-head weights to [H,D,E])
  {
    int nx = cB * cT * cE;
    cvt_f32_bf16_kernel<<<(nx + 255) / 256, 256, 0, stream>>>(xb, x, nx);
    int nw = cH * cD * cE;
    transpose_w_kernel<<<(nw + 255) / 256, 256, 0, stream>>>(Wqt, Wq);
    transpose_w_kernel<<<(nw + 255) / 256, 256, 0, stream>>>(Wkt, Wk);
    transpose_w_kernel<<<(nw + 255) / 256, 256, 0, stream>>>(Wvt, Wv);
    int np = cE * cE;
    cvt_f32_bf16_kernel<<<(np + 255) / 256, 256, 0, stream>>>(Wpb, Wp, np);
  }

  const int strips = cB * cH * (cT / 16);   // 8192 waves, 8 waves/block
  dim3 blk(256);
  dim3 grd(strips / 8);

  // 2) Q/K/V projections (V stored transposed), LDS-staged weights via
  //    async DMA
  qkv_kernel<<<grd, blk, 0, stream>>>(xb, Wqt, Qb, 0);
  qkv_kernel<<<grd, blk, 0, stream>>>(xb, Wkt, Kb, 0);
  qkv_kernel<<<grd, blk, 0, stream>>>(xb, Wvt, Vtb, 1);

  // 3) causal flash attention -> concat-head O [B*T, E]
  attn_kernel<<<grd, blk, 0, stream>>>(Qb, Kb, Vtb, Ob);

  // 4) output projection + bias -> f32
  const int pstrips = (cB * cT / 16) * (cE / 64); // 8192
  proj_kernel<<<dim3(pstrips / 8), blk, 0, stream>>>(Ob, Wpb, bp, y);
}
